// LSTM_31868657336605
// MI455X (gfx1250) — matrix operand order repaired and checked by reading, not compiled, not run on hardware
//
#include <hip/hip_runtime.h>

// Liquid-LSTM scan for MI455X (gfx1250): persistent kernel, weights resident
// in VGPRs as WMMA B-fragments, bf16 v_wmma_f32_16x16x32_bf16 for the
// recurrent GEMM, f32 cell state in registers, activations in LDS.
// Elementwise uses v_exp_f32 + v_rcp_f32 (TRANS) only -- no IEEE divides.
//
// B=2048 T=1024 D_IN=8 SD=2 H=128 Z=130, gates N=512, padded K=160.

#define B_    2048
#define T_    1024
#define DIN   8
#define SDIM  2
#define H_    128
#define Z_    130
#define KP    160           // [x(2) | inp(8) | zero-pad(22) | h(128)]
#define ASTR  168           // LDS act row stride (bf16): 336B = 21*16B, conflict-free
#define MROWS 32            // batch rows per workgroup
#define NWG   (B_ / MROWS)  // 64 workgroups
#define NTHR  256           // 8 waves (wave32); wave id == hidden tile id

typedef __bf16 bf16_t;
typedef __attribute__((ext_vector_type(16))) __bf16 v16bf;
typedef __attribute__((ext_vector_type(8)))  __bf16 v8bf;
typedef __attribute__((ext_vector_type(8)))  float  v8f;

__device__ __forceinline__ float fast_rcp(float x) {
    return __builtin_amdgcn_rcpf(x);          // v_rcp_f32 (TRANS)
}
__device__ __forceinline__ float sigm(float x) {
    return fast_rcp(1.0f + __expf(-x));       // v_exp_f32 + v_rcp_f32
}
__device__ __forceinline__ float tanh_fast(float x) {
    return 1.0f - 2.0f * fast_rcp(__expf(2.0f * x) + 1.0f);
}

__global__ __launch_bounds__(NTHR, 1)
void lnn_lstm_kernel(const float* __restrict__ x_in,   // (B,T,8)
                     const float* __restrict__ tau,    // (B,T,1)
                     const float* __restrict__ z0,     // (1,B,130)
                     const float* __restrict__ c0,     // (1,B,128)
                     const float* __restrict__ W_w,    // (512,10)
                     const float* __restrict__ W_b,    // (512,)
                     const float* __restrict__ U_w,    // (512,128)
                     const int*   __restrict__ lengths,// (B,)
                     float* __restrict__ out)
{
    __shared__ bf16_t act[MROWS * ASTR];   // A matrix: [row][K] bf16, padded stride
    __shared__ bf16_t hstage[MROWS * H_];  // staged h_new (bf16)
    __shared__ float  xstate[MROWS * 2];   // f32 ODE state x

    const int th   = threadIdx.x;
    const int lane = th & 31;
    const int l16  = lane & 15;
    const int kh   = lane >> 4;        // half-wave: selects K sub-chunk / M row group
    const int ht   = th >> 5;          // wave id == hidden tile (16 cols) 0..7
    const int r0   = blockIdx.x * MROWS;

    // ---------------- time-invariant weights -> VGPR B-fragments ----------------
    // B matrix layout (K x N, 16-bit): lane = N column (l16), VGPRs hold
    // contiguous K range [kh*16, kh*16+16) within each 32-K tile.
    v16bf bw[4][5];
    #pragma unroll
    for (int gi = 0; gi < 4; ++gi) {
        const int ncol = gi * H_ + ht * 16 + l16;   // gate output column 0..511
        #pragma unroll
        for (int kt = 0; kt < 5; ++kt) {
            v16bf b;
            #pragma unroll
            for (int e = 0; e < 16; ++e) {
                const int k = kt * 32 + kh * 16 + e;
                float v;
                if (k < SDIM + DIN)      v = W_w[ncol * (SDIM + DIN) + k];
                else if (k < 32)         v = 0.0f;            // zero pad columns
                else                     v = U_w[ncol * H_ + (k - 32)];
                b[e] = (bf16_t)v;
            }
            bw[gi][kt] = b;
        }
    }
    float bias[4];
    #pragma unroll
    for (int gi = 0; gi < 4; ++gi)
        bias[gi] = W_b[gi * H_ + ht * 16 + l16];

    // ---------------- initial state ----------------
    for (int i = th; i < MROWS * ASTR; i += NTHR) act[i] = (bf16_t)0.0f;
    __syncthreads();
    for (int i = th; i < MROWS * H_; i += NTHR) {
        int row = i >> 7, hid = i & 127;
        float hv = z0[(size_t)(r0 + row) * Z_ + SDIM + hid];
        act[row * ASTR + 32 + hid] = (bf16_t)hv;
        hstage[row * H_ + hid]     = (bf16_t)hv;
    }
    if (th < MROWS) {
        float x0 = z0[(size_t)(r0 + th) * Z_ + 0];
        float x1 = z0[(size_t)(r0 + th) * Z_ + 1];
        xstate[th * 2 + 0] = x0;
        xstate[th * 2 + 1] = x1;
        act[th * ASTR + 0] = (bf16_t)x0;
        act[th * ASTR + 1] = (bf16_t)x1;
    }
    // c state: lane-local, matches WMMA C/D layout: row = mt*16 + (kh? j+8 : j)
    float cst[16];
    #pragma unroll
    for (int mt = 0; mt < 2; ++mt)
        #pragma unroll
        for (int j = 0; j < 8; ++j) {
            int row = mt * 16 + (kh ? j + 8 : j);
            cst[mt * 8 + j] = c0[(size_t)(r0 + row) * H_ + ht * 16 + l16];
        }
    __syncthreads();

    const size_t ZROW = (size_t)T_ * Z_;

    // loop-invariant per-thread I/O descriptors
    const int orow  = th >> 3;             // output row this thread owns
    const int og    = th & 7;              // 16-wide hidden column group
    const int ogrow = r0 + orow;
    const int mylen = lengths[ogrow];      // hoisted out of the scan
    const size_t obase = (size_t)ogrow * ZROW;

    // ---------------- sequential scan over T ----------------
    #pragma unroll 1
    for (int t = 0; t < T_; ++t) {
        // Phase A: stream inp_t into act columns [2,10)
        {
            size_t gi = ((size_t)ogrow * T_ + t) * DIN + og;
            act[orow * ASTR + SDIM + og] = (bf16_t)x_in[gi];
            if (t + 1 < T_)
                __builtin_prefetch(&x_in[gi + DIN], 0, 0);   // global_prefetch_b8
        }
        __syncthreads();

        // Phase B: gates = act @ Waug^T + b  (WMMA), then lane-local LSTM cell
        #pragma unroll
        for (int mt = 0; mt < 2; ++mt) {
            v8f acc[4];
            #pragma unroll
            for (int gi = 0; gi < 4; ++gi) {
                v8f c;
                #pragma unroll
                for (int j = 0; j < 8; ++j) c[j] = bias[gi];
                acc[gi] = c;
            }
            #pragma unroll
            for (int kt = 0; kt < 5; ++kt) {
                // A fragment (16x32 bf16): lane holds row m = l16, contiguous K
                // range [kt*32 + kh*16, +16) -> 2x ds_load_b128
                union { v16bf v; v8bf h[2]; } a;
                const bf16_t* rp = &act[(mt * 16 + l16) * ASTR + kt * 32 + kh * 16];
                a.h[0] = *(const v8bf*)(rp);
                a.h[1] = *(const v8bf*)(rp + 8);
                #pragma unroll
                for (int gi = 0; gi < 4; ++gi)
                    acc[gi] = __builtin_amdgcn_wmma_f32_16x16x32_bf16(
                        false, a.v, false, bw[gi][kt], (short)0, acc[gi],
                        false, false);
            }
            #pragma unroll
            for (int j = 0; j < 8; ++j) {
                float iv = sigm(acc[0][j]);
                float fv = sigm(acc[1][j]);
                float gv = tanh_fast(acc[2][j]);
                float ov = sigm(acc[3][j]);
                float cn = fv * cst[mt * 8 + j] + iv * gv;
                cst[mt * 8 + j] = cn;
                float hn = ov * tanh_fast(cn);
                int row = mt * 16 + (kh ? j + 8 : j);
                hstage[row * H_ + ht * 16 + l16] = (bf16_t)hn;
            }
        }
        // tiny 2-state ODE for x (f32, one thread per row)
        if (th < MROWS) {
            float x0 = xstate[th * 2 + 0];
            float x1 = xstate[th * 2 + 1];
            float tt = tau[(size_t)(r0 + th) * T_ + t];
            xstate[th * 2 + 0] = x0 + tt * (1.5f * x0 - 1.5f * x1);
            xstate[th * 2 + 1] = x1 + tt * (x0 * (1.0f / 1.5f));
        }
        __syncthreads();

        // Phase C: commit h/x into act for next step; write masked outputs
        #pragma unroll
        for (int i = 0; i < 2; ++i) {
            int c = th * 2 + i;                 // 512 16B chunks
            int row = c >> 4, cc = c & 15;
            *(v8bf*)&act[row * ASTR + 32 + cc * 8] =
                *(const v8bf*)&hstage[row * H_ + cc * 8];
        }
        if (th < MROWS) {
            act[th * ASTR + 0] = (bf16_t)xstate[th * 2 + 0];
            act[th * ASTR + 1] = (bf16_t)xstate[th * 2 + 1];
        }
        {
            const bool live = t < mylen;
            float* op = out + obase + (size_t)t * Z_;
            if (og == 0) {
                float2 xv;
                xv.x = live ? xstate[orow * 2 + 0] : 0.0f;
                xv.y = live ? xstate[orow * 2 + 1] : 0.0f;
                *(float2*)op = xv;                      // element offset even -> 8B ok
            }
            #pragma unroll
            for (int e = 0; e < 8; ++e) {
                int hid = og * 16 + e * 2;
                float2 hv;
                hv.x = live ? (float)hstage[orow * H_ + hid + 0] : 0.0f;
                hv.y = live ? (float)hstage[orow * H_ + hid + 1] : 0.0f;
                *(float2*)(op + SDIM + hid) = hv;       // SDIM+hid even -> 8B aligned
            }
        }
        // no trailing barrier needed: next-iteration A->B barrier orders
        // phase-C act/hstage traffic against the next WMMA reads/writes.
    }

    // ---------------- final states: z_fin (B,130) then c_fin (B,128) ----------------
    const size_t OUT0 = (size_t)B_ * T_ * Z_;
    const size_t OUT1 = OUT0 + (size_t)B_ * Z_;
    {
        float* zp = out + OUT0 + (size_t)ogrow * Z_;
        if (og == 0) {
            float2 xv;
            xv.x = xstate[orow * 2 + 0];
            xv.y = xstate[orow * 2 + 1];
            *(float2*)zp = xv;
        }
        #pragma unroll
        for (int e = 0; e < 8; ++e) {
            int hid = og * 16 + e * 2;
            float2 hv;
            hv.x = (float)hstage[orow * H_ + hid + 0];
            hv.y = (float)hstage[orow * H_ + hid + 1];
            *(float2*)(zp + SDIM + hid) = hv;
        }
    }
    #pragma unroll
    for (int mt = 0; mt < 2; ++mt)
        #pragma unroll
        for (int j = 0; j < 8; ++j) {
            int row = mt * 16 + (kh ? j + 8 : j);
            out[OUT1 + (size_t)(r0 + row) * H_ + ht * 16 + l16] = cst[mt * 8 + j];
        }
}

extern "C" void kernel_launch(void* const* d_in, const int* in_sizes, int n_in,
                              void* d_out, int out_size, void* d_ws, size_t ws_size,
                              hipStream_t stream) {
    (void)in_sizes; (void)n_in; (void)d_ws; (void)ws_size; (void)out_size;
    const float* x_in = (const float*)d_in[0];
    const float* tau  = (const float*)d_in[1];
    const float* z0   = (const float*)d_in[2];
    const float* c0   = (const float*)d_in[3];
    const float* W_w  = (const float*)d_in[4];
    const float* W_b  = (const float*)d_in[5];
    const float* U_w  = (const float*)d_in[6];
    const int*   len  = (const int*)d_in[7];
    float* out = (float*)d_out;
    lnn_lstm_kernel<<<NWG, NTHR, 0, stream>>>(x_in, tau, z0, c0, W_w, W_b, U_w,
                                              len, out);
}
